// VectorQuantizer_50105088475483
// MI455X (gfx1250) — matrix operand order repaired and checked by reading, not compiled, not run on hardware
//
#include <hip/hip_runtime.h>

#define NUM_CODES 1024
#define CODE_DIM  64
#define B_ROWS    262144

typedef __attribute__((ext_vector_type(16))) __bf16 v16bf;
typedef __attribute__((ext_vector_type(8)))  float  v8f;

// ---- d_out flat layout (floats), reference tuple order ----
#define OFF_ZQ   0
#define OFF_IDX  (B_ROWS * CODE_DIM)            // 16777216
#define OFF_LOSS (OFF_IDX + B_ROWS)             // 17039360
#define OFF_CS   (OFF_LOSS + 1)                 // 17039361
#define OFF_ES   (OFF_CS + NUM_CODES)           // 17040385
#define OFF_W    (OFF_ES + NUM_CODES * CODE_DIM)// 17105921

// ---- workspace layout (bytes) ----
#define WS_W2_OFF   0                    // 1024 floats
#define WS_WHI_OFF  4096                 // 65536 ushort
#define WS_WLO_OFF  (4096 + 131072)      // 65536 ushort
#define WS_N_OFF    (4096 + 262144)      // 1 float

#define ONE_MINUS_DECAY ((float)(1.0 - 0.99))   // == f32(0.01), matches JAX
#define DECAY 0.99f

struct U32x8 { unsigned w[8]; };   // 32 bytes -> v16bf
struct U4x2  { uint4 a, b; };      // 32 bytes -> v16bf

__device__ __forceinline__ unsigned short f2bf(float f) {
  unsigned u = __float_as_uint(f);
  u += 0x7FFFu + ((u >> 16) & 1u);     // round-to-nearest-even
  return (unsigned short)(u >> 16);
}
__device__ __forceinline__ float bf2f(unsigned short b) {
  return __uint_as_float(((unsigned)b) << 16);
}

__device__ __forceinline__ void wait_async0() {
#if __has_builtin(__builtin_amdgcn_s_wait_asynccnt)
  __builtin_amdgcn_s_wait_asynccnt(0);
#else
  asm volatile("s_wait_asynccnt 0x0" ::: "memory");
#endif
}

// Async DMA: 16 bytes global -> LDS, tracked with ASYNCcnt (no VGPR round-trip).
__device__ __forceinline__ void async_g2l_b128(unsigned lds_addr, unsigned long long gaddr) {
  asm volatile("global_load_async_to_lds_b128 %0, %1, off"
               :: "v"(lds_addr), "v"(gaddr) : "memory");
}

#define WMMA_BF16(A, Bm, C) \
  __builtin_amdgcn_wmma_f32_16x16x32_bf16(false, (A), false, (Bm), (short)0, (C), false, false)

// ---------------- init: split weight into bf16 hi/lo, w2, EMA bases ----------------
__global__ void vq_init(const float* __restrict__ weight,
                        const float* __restrict__ ema_cs,
                        const float* __restrict__ ema_es,
                        unsigned short* __restrict__ whi,
                        unsigned short* __restrict__ wlo,
                        float* __restrict__ w2,
                        float* __restrict__ wsn,
                        float* __restrict__ out) {
  int i = blockIdx.x * blockDim.x + threadIdx.x;
  if (i < NUM_CODES * CODE_DIM) {
    float x = weight[i];
    unsigned short h = f2bf(x);
    unsigned short l = f2bf(x - bf2f(h));
    whi[i] = h; wlo[i] = l;
    out[OFF_ES + i] = DECAY * ema_es[i];     // kernel1 atomics add (1-d)*sum(z)
  }
  if (i < NUM_CODES) {
    float s = 0.f;
    #pragma unroll 8
    for (int d = 0; d < CODE_DIM; ++d) { float w = weight[i * CODE_DIM + d]; s = fmaf(w, w, s); }
    w2[i] = s;
    out[OFF_CS + i] = DECAY * ema_cs[i];     // kernel1 atomics add (1-d)*count
  }
  if (i == 0) { out[OFF_LOSS] = 0.f; *wsn = 0.f; }
}

// ---------------- main: WMMA distances + argmin + gather + atomics ----------------
__global__ __launch_bounds__(256) void
vq_main(const float* __restrict__ z,
        const float* __restrict__ weight,
        const unsigned short* __restrict__ whi,
        const unsigned short* __restrict__ wlo,
        const float* __restrict__ w2,
        float* __restrict__ out) {
  __shared__ __align__(16) unsigned short sh_hi[2][16 * 72];  // padded rows: no bank conflicts
  __shared__ __align__(16) unsigned short sh_lo[2][16 * 72];
  __shared__ float sh_w2[NUM_CODES];

  const int t    = threadIdx.x;
  const int lane = t & 31;
  const int wv   = t >> 5;        // wave id 0..7
  const int half = lane >> 4;     // 0|1
  const int l16  = lane & 15;

  // ---- build A operands (z rows) in a 16x32 A layout with contiguous k per lane half, bf16 hi/lo split ----
  v16bf a_hi[2][2], a_lo[2][2];
  const int kb = half * 16;       // lanes 0-15: K 0..15 ; lanes 16-31: K 16..31
  #pragma unroll
  for (int mt = 0; mt < 2; ++mt) {
    const int arow = blockIdx.x * 256 + wv * 32 + mt * 16 + l16;
    const float* zr = z + (size_t)arow * CODE_DIM;
    #pragma unroll
    for (int h = 0; h < 2; ++h) {
      float buf[16];
      *(float4*)(&buf[0])  = *(const float4*)(zr + 32 * h + kb);
      *(float4*)(&buf[4])  = *(const float4*)(zr + 32 * h + kb + 4);
      *(float4*)(&buf[8])  = *(const float4*)(zr + 32 * h + kb + 8);
      *(float4*)(&buf[12]) = *(const float4*)(zr + 32 * h + kb + 12);
      U32x8 ph, pl;
      #pragma unroll
      for (int w = 0; w < 8; ++w) {
        float x0 = buf[2 * w], x1 = buf[2 * w + 1];
        unsigned short h0 = f2bf(x0), h1 = f2bf(x1);
        unsigned short l0 = f2bf(x0 - bf2f(h0)), l1 = f2bf(x1 - bf2f(h1));
        ph.w[w] = (unsigned)h0 | ((unsigned)h1 << 16);
        pl.w[w] = (unsigned)l0 | ((unsigned)l1 << 16);
      }
      a_hi[mt][h] = __builtin_bit_cast(v16bf, ph);
      a_lo[mt][h] = __builtin_bit_cast(v16bf, pl);
    }
  }

  // preload w2 into LDS
  for (int j = t; j < NUM_CODES; j += 256) sh_w2[j] = w2[j];

  // Async stage of one 16x64 bf16 tile (hi + lo) into LDS buffer b.
  // Waves 0-3 stage the hi table, waves 4-7 the lo table: 128 threads x 16B = 2KB each.
  auto stage_async = [&](int nt, int b) {
    const int tbl = t >> 7;              // wave-uniform: 0 = hi, 1 = lo
    const int j   = t & 127;
    const int row = j >> 3;              // 16 rows
    const int col = (j & 7) * 8;         // 8 chunks of 8 ushorts (16B)
    const unsigned long long gaddr =
        (unsigned long long)(tbl ? (const void*)wlo : (const void*)whi) +
        (unsigned long long)(((nt * 16 + row) * CODE_DIM + col) * 2);
    unsigned short* dst = tbl ? &sh_lo[b][row * 72 + col] : &sh_hi[b][row * 72 + col];
    async_g2l_b128((unsigned)(uintptr_t)dst, gaddr);
  };

  float best[2][8];
  int   bidx[2][8];
  #pragma unroll
  for (int mt = 0; mt < 2; ++mt)
    #pragma unroll
    for (int v = 0; v < 8; ++v) { best[mt][v] = 3.4028235e38f; bidx[mt][v] = 0; }

  stage_async(0, 0);

  for (int nt = 0; nt < 64; ++nt) {
    wait_async0();        // my async LDS writes (issued last iteration) are done
    __syncthreads();      // everyone's are -> buffer (nt&1) is ready
    if (nt + 1 < 64) stage_async(nt + 1, (nt + 1) & 1);
    const int b = nt & 1;

    // B operands: 32x16 bf16 layout — lanes 0-15 K=0..15, lanes 16-31 K=16..31, N = lane%16
    v16bf bhi[2], blo[2];
    #pragma unroll
    for (int h = 0; h < 2; ++h) {
      const unsigned short* ph = &sh_hi[b][l16 * 72 + half * 16 + 32 * h];
      const unsigned short* pl = &sh_lo[b][l16 * 72 + half * 16 + 32 * h];
      U4x2 qh { ((const uint4*)ph)[0], ((const uint4*)ph)[1] };
      U4x2 ql { ((const uint4*)pl)[0], ((const uint4*)pl)[1] };
      bhi[h] = __builtin_bit_cast(v16bf, qh);
      blo[h] = __builtin_bit_cast(v16bf, ql);
    }
    const float w2v = sh_w2[nt * 16 + l16];
    const int cand = nt * 16 + l16;

    #pragma unroll
    for (int mt = 0; mt < 2; ++mt) {
      v8f acc = {};
      // bf16x3: small terms first, then hi*hi — f32 accumulate inside WMMA
      acc = WMMA_BF16(a_lo[mt][0], bhi[0], acc);
      acc = WMMA_BF16(a_hi[mt][0], blo[0], acc);
      acc = WMMA_BF16(a_hi[mt][0], bhi[0], acc);
      acc = WMMA_BF16(a_lo[mt][1], bhi[1], acc);
      acc = WMMA_BF16(a_hi[mt][1], blo[1], acc);
      acc = WMMA_BF16(a_hi[mt][1], bhi[1], acc);
      #pragma unroll
      for (int v = 0; v < 8; ++v) {
        float dist = fmaf(-2.0f, acc[v], w2v);   // ||z||^2 cancels in argmin
        if (dist < best[mt][v]) { best[mt][v] = dist; bidx[mt][v] = cand; }
      }
    }
  }

  // ---- argmin across the 16 lanes sharing each row (tie -> smaller index) ----
  float lossAcc = 0.f;
  float* out_es = out + OFF_ES;
  float* out_cs = out + OFF_CS;

  #pragma unroll
  for (int mt = 0; mt < 2; ++mt) {
    #pragma unroll
    for (int v = 0; v < 8; ++v) {
      #pragma unroll
      for (int m = 1; m < 16; m <<= 1) {
        float od = __shfl_xor(best[mt][v], m, 16);
        int   oi = __shfl_xor(bidx[mt][v], m, 16);
        if (od < best[mt][v] || (od == best[mt][v] && oi < bidx[mt][v])) {
          best[mt][v] = od; bidx[mt][v] = oi;
        }
      }
    }
    const int mbase = blockIdx.x * 256 + wv * 32 + mt * 16;
    #pragma unroll
    for (int r = 0; r < 16; ++r) {
      const int idx  = __shfl(bidx[mt][r & 7], (r < 8) ? 0 : 16, 32); // uniform
      const int rowg = mbase + r;
      const int d    = lane * 2;
      float2 wv2 = *(const float2*)(weight + (size_t)idx * CODE_DIM + d);
      float2 zv2 = *(const float2*)(z + (size_t)rowg * CODE_DIM + d);
      *(float2*)(out + OFF_ZQ + (size_t)rowg * CODE_DIM + d) = wv2;   // z_q_st == z_q
      float dx = zv2.x - wv2.x, dy = zv2.y - wv2.y;
      lossAcc = fmaf(dx, dx, lossAcc);
      lossAcc = fmaf(dy, dy, lossAcc);
      atomicAdd(out_es + idx * CODE_DIM + d,     ONE_MINUS_DECAY * zv2.x);
      atomicAdd(out_es + idx * CODE_DIM + d + 1, ONE_MINUS_DECAY * zv2.y);
      if (lane == 0) {
        out[OFF_IDX + rowg] = (float)idx;
        atomicAdd(out_cs + idx, ONE_MINUS_DECAY);
      }
    }
  }

  #pragma unroll
  for (int m = 16; m >= 1; m >>= 1) lossAcc += __shfl_xor(lossAcc, m, 32);
  if (lane == 0) atomicAdd(out + OFF_LOSS, lossAcc);
}

// ---------------- finalize: n = sum(new_cs), scale loss ----------------
__global__ void vq_reduce_n(float* __restrict__ out, float* __restrict__ wsn) {
  __shared__ float red[1024];
  int t = threadIdx.x;
  red[t] = out[OFF_CS + t];
  __syncthreads();
  for (int s = 512; s > 0; s >>= 1) { if (t < s) red[t] += red[t + s]; __syncthreads(); }
  if (t == 0) {
    *wsn = red[0];
    out[OFF_LOSS] = 0.25f * out[OFF_LOSS] / (float)(B_ROWS * CODE_DIM);
  }
}

// ---------------- finalize: new_weight = new_es / smoothed ----------------
__global__ void vq_finalize(float* __restrict__ out, const float* __restrict__ wsn) {
  int i = blockIdx.x * blockDim.x + threadIdx.x;
  if (i >= NUM_CODES * CODE_DIM) return;
  int k = i >> 6;
  float n  = *wsn;
  float cs = out[OFF_CS + k];
  float smoothed = (cs + 1e-5f) / (n + (float)NUM_CODES * 1e-5f) * n;
  out[OFF_W + i] = out[OFF_ES + i] / smoothed;
}

extern "C" void kernel_launch(void* const* d_in, const int* in_sizes, int n_in,
                              void* d_out, int out_size, void* d_ws, size_t ws_size,
                              hipStream_t stream) {
  const float* z       = (const float*)d_in[0];
  const float* weight  = (const float*)d_in[1];
  const float* ema_cs  = (const float*)d_in[2];
  const float* ema_es  = (const float*)d_in[3];
  float* out = (float*)d_out;

  float*          w2  = (float*)((char*)d_ws + WS_W2_OFF);
  unsigned short* whi = (unsigned short*)((char*)d_ws + WS_WHI_OFF);
  unsigned short* wlo = (unsigned short*)((char*)d_ws + WS_WLO_OFF);
  float*          wsn = (float*)((char*)d_ws + WS_N_OFF);

  vq_init    <<<256, 256, 0, stream>>>(weight, ema_cs, ema_es, whi, wlo, w2, wsn, out);
  vq_main    <<<B_ROWS / 256, 256, 0, stream>>>(z, weight, whi, wlo, w2, out);
  vq_reduce_n<<<1, 1024, 0, stream>>>(out, wsn);
  vq_finalize<<<(NUM_CODES * CODE_DIM) / 256, 256, 0, stream>>>(out, wsn);
}
